// VAE_Decoder_64372969832521
// MI455X (gfx1250) — matrix operand order repaired and checked by reading, not compiled, not run on hardware
//
#include <hip/hip_runtime.h>
#include <hip/hip_bf16.h>
#include <math.h>

typedef __attribute__((ext_vector_type(16))) _Float16 v16h;
typedef __attribute__((ext_vector_type(8)))  float    v8f;

// Problem constants (match reference)
constexpr int Hn = 12;     // hidden size
constexpr int Ln = 120;    // LSTM layers
constexpr int Bn = 128;    // batch
constexpr int Tn = 256;    // time steps
constexpr int Cn = 6;      // conv channels
constexpr int Kn = 5;      // convtranspose kernel
constexpr int Gn = 48;     // 4*Hn gate width
constexpr int RING = 8;    // ring depth (timesteps) for layer pipeline

__device__ __forceinline__ int ldv(const int* p) { return *(volatile const int*)p; }
__device__ __forceinline__ void stv(int* p, int v) { *(volatile int*)p = v; }
__device__ __forceinline__ float ldvf(const float* p) { return *(volatile const float*)p; }

__device__ __forceinline__ float sigmoidf_(float x) { return 1.0f / (1.0f + __expf(-x)); }

// ---------------------------------------------------------------------------
// 0. zero the pipeline progress flags (re-run every call: deterministic)
// ---------------------------------------------------------------------------
__global__ void init_flags_kernel(int* flags, int n) {
    int i = blockIdx.x * blockDim.x + threadIdx.x;
    if (i < n) flags[i] = 0;
}

// ---------------------------------------------------------------------------
// 1. h = relu(relu(x@W1^T+b1)@W2^T+b2); store transposed to [T,B,H]
// ---------------------------------------------------------------------------
__global__ __launch_bounds__(256) void pre_kernel(
    const float* __restrict__ x, const float* __restrict__ w1, const float* __restrict__ b1,
    const float* __restrict__ w2, const float* __restrict__ b2, float* __restrict__ xs) {
    int idx = blockIdx.x * blockDim.x + threadIdx.x;   // over B*T rows
    if (idx >= Bn * Tn) return;
    int b = idx / Tn, t = idx % Tn;
    float xin[Hn], h1[Hn];
    #pragma unroll
    for (int i = 0; i < Hn; ++i) xin[i] = x[(b * Tn + t) * Hn + i];
    #pragma unroll
    for (int o = 0; o < Hn; ++o) {
        float s = b1[o];
        #pragma unroll
        for (int i = 0; i < Hn; ++i) s += xin[i] * w1[o * Hn + i];
        h1[o] = fmaxf(s, 0.0f);
    }
    #pragma unroll
    for (int o = 0; o < Hn; ++o) {
        float s = b2[o];
        #pragma unroll
        for (int i = 0; i < Hn; ++i) s += h1[i] * w2[o * Hn + i];
        xs[(t * Bn + b) * Hn + o] = fmaxf(s, 0.0f);
    }
}

// ---------------------------------------------------------------------------
// 2. Wavefront LSTM: block = layer, 8 waves = 8 batch tiles of 16.
//    gates[16,48] = A[16,32] x B[32,48] via 3x v_wmma_f32_16x16x32_f16.
// ---------------------------------------------------------------------------
__global__ __launch_bounds__(256) void lstm_wavefront_kernel(
    const float* __restrict__ xs,        // [T,B,H]  input to layer 0
    const float* __restrict__ wih_all,   // [L,48,12]
    const float* __restrict__ whh_all,   // [L,48,12]
    const float* __restrict__ bih_all,   // [L,48]
    const float* __restrict__ bhh_all,   // [L,48]
    float* __restrict__ ring,            // [L,RING,B,H]
    float* __restrict__ hidden,          // [L,B,H] final h per layer
    int* produced, int* consumed)        // [L] progress counters
{
    const int l    = blockIdx.x;           // layer
    const int wave = threadIdx.x >> 5;     // 0..7 -> batch tile
    const int lane = threadIdx.x & 31;
    const int brow0 = wave * 16;

    __shared__ _Float16 xh[8][16][32];     // per-wave A staging: [x | h | pad]
    __shared__ float    gbuf[8][16][48];   // per-wave gate tile
    __shared__ float    cbuf[8][16][12];   // per-wave cell state
    __shared__ float    bias[48];

    // --- init shared state -------------------------------------------------
    if (threadIdx.x < Gn)
        bias[threadIdx.x] = bih_all[l * Gn + threadIdx.x] + bhh_all[l * Gn + threadIdx.x];
    for (int e = lane; e < 16 * 12; e += 32) cbuf[wave][e / 12][e % 12] = 0.0f;
    for (int e = lane; e < 16 * 32; e += 32) xh[wave][e / 32][e % 32] = (_Float16)0.0f;

    // --- build B tiles in registers (held for all 256 steps) ---------------
    // B operand, contiguous K order used here (A matches): lane%16 = column N, lanes16-31 carry K=16..31,
    // VGPR j holds K = kb+2j, kb+2j+1  => half slot p maps to K = kb + p.
    const float* wih = wih_all + l * Gn * Hn;
    const float* whh = whh_all + l * Gn * Hn;
    const int coln = lane & 15;
    const int kbB  = (lane >> 4) ? 16 : 0;
    v16h bt0, bt1, bt2;
    #pragma unroll
    for (int p = 0; p < 16; ++p) {
        int K = kbB + p;
        float v0 = 0.f, v1 = 0.f, v2 = 0.f;
        if (K < 12) {
            v0 = wih[(0 * 16 + coln) * Hn + K];
            v1 = wih[(1 * 16 + coln) * Hn + K];
            v2 = wih[(2 * 16 + coln) * Hn + K];
        } else if (K < 24) {
            v0 = whh[(0 * 16 + coln) * Hn + (K - 12)];
            v1 = whh[(1 * 16 + coln) * Hn + (K - 12)];
            v2 = whh[(2 * 16 + coln) * Hn + (K - 12)];
        }
        bt0[p] = (_Float16)v0; bt1[p] = (_Float16)v1; bt2[p] = (_Float16)v2;
    }
    __syncthreads();

    const int m    = lane & 15;            // A row (batch within tile)
    const int kbA  = (lane >> 4) ? 16 : 0; // A K-base (contiguous, matches B)
    const int rowoff = (lane >> 4) ? 8 : 0;

    for (int t = 0; t < Tn; ++t) {
        // ---- wait for producer layer --------------------------------------
        if (l > 0) {
            if (threadIdx.x == 0)
                while (ldv(produced + (l - 1)) < t + 1) __builtin_amdgcn_s_sleep(1);
            __syncthreads();
            __threadfence();
        }
        // ---- stage x_t tile into xh columns 0..11 -------------------------
        const float* src = (l == 0)
            ? (xs + (size_t)t * Bn * Hn)
            : (ring + (((size_t)(l - 1) * RING + (t & (RING - 1))) * Bn * Hn));
        for (int e = lane; e < 16 * 12; e += 32) {
            int r = e / 12, cc = e % 12;
            float v = (l == 0) ? src[(brow0 + r) * Hn + cc]
                               : ldvf(src + (brow0 + r) * Hn + cc);
            xh[wave][r][cc] = (_Float16)v;
        }
        __syncthreads();
        if (l > 0 && threadIdx.x == 0) stv(consumed + l, t + 1);  // slot drained

        // ---- gather A operand (16-bit A 16x32 layout) ---------------------
        v16h a;
        #pragma unroll
        for (int p = 0; p < 16; ++p) {
            int K = kbA + p;
            a[p] = xh[wave][m][K];
        }
        // ---- 3 WMMAs: gates[16,48] ----------------------------------------
        v8f c0 = {}, c1 = {}, c2 = {};
        c0 = __builtin_amdgcn_wmma_f32_16x16x32_f16(false, a, false, bt0, (short)0, c0, false, false);
        c1 = __builtin_amdgcn_wmma_f32_16x16x32_f16(false, a, false, bt1, (short)0, c1, false, false);
        c2 = __builtin_amdgcn_wmma_f32_16x16x32_f16(false, a, false, bt2, (short)0, c2, false, false);
        // ---- scatter C (lane=column, VGPR=row) ----------------------------
        #pragma unroll
        for (int r = 0; r < 8; ++r) {
            gbuf[wave][rowoff + r][coln]      = c0[r];
            gbuf[wave][rowoff + r][16 + coln] = c1[r];
            gbuf[wave][rowoff + r][32 + coln] = c2[r];
        }

        // ---- back-pressure: don't overwrite a ring slot still in flight ---
        if (l < Ln - 1 && t >= RING) {
            if (threadIdx.x == 0)
                while (ldv(consumed + (l + 1)) < t - RING + 1) __builtin_amdgcn_s_sleep(1);
        }
        __syncthreads();

        // ---- elementwise LSTM cell ----------------------------------------
        float* dst = ring + (((size_t)l * RING + (t & (RING - 1))) * Bn * Hn);
        for (int e = lane; e < 16 * 12; e += 32) {
            int r = e / 12, cc = e % 12;
            float ig = gbuf[wave][r][cc]      + bias[cc];
            float fg = gbuf[wave][r][12 + cc] + bias[12 + cc];
            float gg = gbuf[wave][r][24 + cc] + bias[24 + cc];
            float og = gbuf[wave][r][36 + cc] + bias[36 + cc];
            float cn = sigmoidf_(fg) * cbuf[wave][r][cc] + sigmoidf_(ig) * tanhf(gg);
            float hn = sigmoidf_(og) * tanhf(cn);
            cbuf[wave][r][cc] = cn;
            xh[wave][r][12 + cc] = (_Float16)hn;          // next step's A operand
            if (l < Ln - 1) dst[(brow0 + r) * Hn + cc] = hn;
            if (t == Tn - 1) hidden[((size_t)l * Bn + brow0 + r) * Hn + cc] = hn;
        }
        __threadfence();
        __syncthreads();
        if (threadIdx.x == 0) stv(produced + l, t + 1);
    }
}

// ---------------------------------------------------------------------------
// 3. attention projection: [B,L,C] = relu(hidden^T @ attn_w^T + b) -> [B,C,L]
// ---------------------------------------------------------------------------
__global__ __launch_bounds__(256) void attn_kernel(
    const float* __restrict__ hidden, const float* __restrict__ w,
    const float* __restrict__ b, float* __restrict__ out) {
    int idx = blockIdx.x * blockDim.x + threadIdx.x;
    if (idx >= Bn * Cn * Ln) return;
    int bb = idx / (Cn * Ln);
    int c  = (idx / Ln) % Cn;
    int l  = idx % Ln;
    float s = b[c];
    #pragma unroll
    for (int h = 0; h < Hn; ++h) s += hidden[((size_t)l * Bn + bb) * Hn + h] * w[c * Hn + h];
    out[((size_t)bb * Cn + c) * Ln + l] = fmaxf(s, 0.0f);
}

// ---------------------------------------------------------------------------
// 4. ConvTranspose1d (stride 1): y[b,co,j] = bias + sum_ci,k w[ci,co,k]*x[b,ci,j-k]
// ---------------------------------------------------------------------------
__global__ __launch_bounds__(256) void deconv_kernel(
    const float* __restrict__ x, const float* __restrict__ w,
    const float* __restrict__ bias, float* __restrict__ y, int Lin) {
    int Lout = Lin + Kn - 1;
    int idx = blockIdx.x * blockDim.x + threadIdx.x;
    if (idx >= Bn * Cn * Lout) return;
    int b  = idx / (Cn * Lout);
    int co = (idx / Lout) % Cn;
    int j  = idx % Lout;
    float s = bias[co];
    for (int ci = 0; ci < Cn; ++ci) {
        #pragma unroll
        for (int k = 0; k < Kn; ++k) {
            int i = j - k;
            if (i >= 0 && i < Lin)
                s += w[(ci * Cn + co) * Kn + k] * x[((size_t)b * Cn + ci) * Lin + i];
        }
    }
    y[((size_t)b * Cn + co) * Lout + j] = s;
}

// ---------------------------------------------------------------------------
// 5. BatchNorm stats (training mode, biased var): one block per channel
// ---------------------------------------------------------------------------
__global__ __launch_bounds__(256) void stats_kernel(
    const float* __restrict__ x, int Len, float* __restrict__ stats) {
    int c = blockIdx.x;
    __shared__ float ssum[256], ssq[256];
    float s = 0.f, q = 0.f;
    for (int e = threadIdx.x; e < Bn * Len; e += 256) {
        int b = e / Len, j = e % Len;
        float v = x[((size_t)b * Cn + c) * Len + j];
        s += v; q += v * v;
    }
    ssum[threadIdx.x] = s; ssq[threadIdx.x] = q;
    __syncthreads();
    for (int st = 128; st > 0; st >>= 1) {
        if (threadIdx.x < st) {
            ssum[threadIdx.x] += ssum[threadIdx.x + st];
            ssq[threadIdx.x]  += ssq[threadIdx.x + st];
        }
        __syncthreads();
    }
    if (threadIdx.x == 0) {
        float n = (float)(Bn * Len);
        float mean = ssum[0] / n;
        float var  = ssq[0] / n - mean * mean;
        stats[c]      = mean;
        stats[Cn + c] = rsqrtf(var + 1e-5f);
    }
}

// ---------------------------------------------------------------------------
// 6. BN normalize + relu (in place)
// ---------------------------------------------------------------------------
__global__ __launch_bounds__(256) void bnrelu_kernel(
    float* __restrict__ x, const float* __restrict__ stats,
    const float* __restrict__ g, const float* __restrict__ b, int Len) {
    int idx = blockIdx.x * blockDim.x + threadIdx.x;
    if (idx >= Bn * Cn * Len) return;
    int c = (idx / Len) % Cn;
    float v = (x[idx] - stats[c]) * stats[Cn + c] * g[c] + b[c];
    x[idx] = fmaxf(v, 0.0f);
}

// ---------------------------------------------------------------------------
// 7. mu / softplus(sig) heads -> d_out (mu flat, then sig flat)
// ---------------------------------------------------------------------------
__global__ __launch_bounds__(256) void head_kernel(
    const float* __restrict__ ang, const float* __restrict__ mu_w,
    const float* __restrict__ mu_b, const float* __restrict__ sig_w,
    const float* __restrict__ sig_b, float* __restrict__ out, int Len) {
    int idx = blockIdx.x * blockDim.x + threadIdx.x;
    if (idx >= Bn * Cn * Len) return;
    int b = idx / (Cn * Len);
    int c = (idx / Len) % Cn;
    int j = idx % Len;
    float sm = mu_b[c], sg = sig_b[c];
    #pragma unroll
    for (int ci = 0; ci < Cn; ++ci) {
        float v = ang[((size_t)b * Cn + ci) * Len + j];
        sm += mu_w[c * Cn + ci] * v;
        sg += sig_w[c * Cn + ci] * v;
    }
    float sp = (sg > 20.0f) ? sg : log1pf(__expf(sg));
    out[idx] = sm;
    out[(size_t)Bn * Cn * Len + idx] = sp;
}

// ---------------------------------------------------------------------------
extern "C" void kernel_launch(void* const* d_in, const int* in_sizes, int n_in,
                              void* d_out, int out_size, void* d_ws, size_t ws_size,
                              hipStream_t stream) {
    const float* x      = (const float*)d_in[0];
    const float* lin1_w = (const float*)d_in[1];
    const float* lin1_b = (const float*)d_in[2];
    const float* lin2_w = (const float*)d_in[3];
    const float* lin2_b = (const float*)d_in[4];
    const float* wih    = (const float*)d_in[5];
    const float* whh    = (const float*)d_in[6];
    const float* bih    = (const float*)d_in[7];
    const float* bhh    = (const float*)d_in[8];
    const float* attn_w = (const float*)d_in[9];
    const float* attn_b = (const float*)d_in[10];
    const float* dca_w  = (const float*)d_in[11];
    const float* dca_b  = (const float*)d_in[12];
    const float* dan_w  = (const float*)d_in[13];
    const float* dan_b  = (const float*)d_in[14];
    const float* bn1_g  = (const float*)d_in[15];
    const float* bn1_b  = (const float*)d_in[16];
    const float* bn2_g  = (const float*)d_in[17];
    const float* bn2_b  = (const float*)d_in[18];
    const float* mu_w   = (const float*)d_in[19];
    const float* mu_b   = (const float*)d_in[20];
    const float* sig_w  = (const float*)d_in[21];
    const float* sig_b  = (const float*)d_in[22];

    // workspace carve-up (floats)
    float* ws   = (float*)d_ws;
    float* xs     = ws;                                   // T*B*H      = 393216
    float* ring   = xs     + (size_t)Tn * Bn * Hn;        // L*RING*B*H = 1474560
    float* hidden = ring   + (size_t)Ln * RING * Bn * Hn; // L*B*H      = 184320
    float* attn_o = hidden + (size_t)Ln * Bn * Hn;        // B*C*120    = 92160
    float* acc    = attn_o + (size_t)Bn * Cn * Ln;        // B*C*124    = 95232
    float* ang    = acc    + (size_t)Bn * Cn * (Ln + 4);  // B*C*128    = 98304
    float* stats  = ang    + (size_t)Bn * Cn * (Ln + 8);  // 24
    int*   flags  = (int*)(stats + 24);                   // produced[L], consumed[L]
    int* produced = flags;
    int* consumed = flags + Ln;

    // 0. reset pipeline flags
    init_flags_kernel<<<1, 256, 0, stream>>>(flags, 2 * Ln);
    // 1. input linears -> [T,B,H]
    pre_kernel<<<(Bn * Tn + 255) / 256, 256, 0, stream>>>(x, lin1_w, lin1_b, lin2_w, lin2_b, xs);
    // 2. layer-wavefront LSTM (1 block per layer, ring-buffered pipeline)
    lstm_wavefront_kernel<<<Ln, 256, 0, stream>>>(xs, wih, whh, bih, bhh,
                                                  ring, hidden, produced, consumed);
    // 3. attention projection -> [B,C,120]
    attn_kernel<<<(Bn * Cn * Ln + 255) / 256, 256, 0, stream>>>(hidden, attn_w, attn_b, attn_o);
    // 4-6. deconv1 + BN1 + relu
    deconv_kernel<<<(Bn * Cn * (Ln + 4) + 255) / 256, 256, 0, stream>>>(attn_o, dca_w, dca_b, acc, Ln);
    stats_kernel<<<Cn, 256, 0, stream>>>(acc, Ln + 4, stats);
    bnrelu_kernel<<<(Bn * Cn * (Ln + 4) + 255) / 256, 256, 0, stream>>>(acc, stats, bn1_g, bn1_b, Ln + 4);
    // deconv2 + BN2 + relu
    deconv_kernel<<<(Bn * Cn * (Ln + 8) + 255) / 256, 256, 0, stream>>>(acc, dan_w, dan_b, ang, Ln + 4);
    stats_kernel<<<Cn, 256, 0, stream>>>(ang, Ln + 8, stats + 12);
    bnrelu_kernel<<<(Bn * Cn * (Ln + 8) + 255) / 256, 256, 0, stream>>>(ang, stats + 12, bn2_g, bn2_b, Ln + 8);
    // 7. heads
    head_kernel<<<(Bn * Cn * (Ln + 8) + 255) / 256, 256, 0, stream>>>(
        ang, mu_w, mu_b, sig_w, sig_b, (float*)d_out, Ln + 8);
}
